// SimpleSelectiveSSM_49031346651480
// MI455X (gfx1250) — matrix operand order repaired and checked by reading, not compiled, not run on hardware
//
#include <hip/hip_runtime.h>
#include <hip/hip_bf16.h>
#include <cstddef>
#include <cstdint>

typedef __bf16 bf16;
typedef __attribute__((ext_vector_type(16))) __bf16 v16bf;
typedef __attribute__((ext_vector_type(8)))  __bf16 v8bf;
typedef __attribute__((ext_vector_type(8)))  float  v8f;

#define Bb 8
#define Tt 4096
#define Dd 2048
#define Ss 128
#define Mrows (Bb * Tt)          // 32768
#define N3 (3 * Ss)              // 384
#define TILE (128 * 32)          // halves per LDS tile buffer
#define CHUNKS 8
#define CLEN (Tt / CHUNKS)       // 512

#define USE_ASYNC_LDS 1

// ---------------------------------------------------------------------------
// Stage 1: fused LayerNorm -> u(bf16), plus bf16 copy of x for skip GEMM
// ---------------------------------------------------------------------------
__global__ void ln_kernel(const float* __restrict__ x,
                          const float* __restrict__ gamma,
                          const float* __restrict__ beta,
                          bf16* __restrict__ ubf,
                          bf16* __restrict__ xbf)
{
    const int row = blockIdx.x;                 // 0..Mrows-1
    const int t   = threadIdx.x;                // 256 threads
    const float* xr = x + (size_t)row * Dd;

    float vals[8];
    float s = 0.f, s2 = 0.f;
#pragma unroll
    for (int i = 0; i < 8; ++i) {
        float v = xr[t + i * 256];
        vals[i] = v;
        s  += v;
        s2 += v * v;
    }

    __shared__ float rs[256];
    __shared__ float rq[256];
    rs[t] = s; rq[t] = s2;
    __syncthreads();
#pragma unroll
    for (int st = 128; st > 0; st >>= 1) {
        if (t < st) { rs[t] += rs[t + st]; rq[t] += rq[t + st]; }
        __syncthreads();
    }
    const float mu   = rs[0] * (1.f / Dd);
    const float var  = rq[0] * (1.f / Dd) - mu * mu;
    const float rstd = rsqrtf(var + 1e-5f);

    bf16* ur = ubf + (size_t)row * Dd;
    bf16* xb = xbf + (size_t)row * Dd;
#pragma unroll
    for (int i = 0; i < 8; ++i) {
        int col = t + i * 256;
        float u = (vals[i] - mu) * rstd * gamma[col] + beta[col];
        ur[col] = (bf16)u;
        xb[col] = (bf16)vals[i];
    }
}

// ---------------------------------------------------------------------------
// fp32 -> bf16 weight conversion
// ---------------------------------------------------------------------------
__global__ void cvt_bf16_kernel(const float* __restrict__ in,
                                bf16* __restrict__ out, size_t n)
{
    size_t i = (size_t)blockIdx.x * blockDim.x + threadIdx.x;
    if (i < n) out[i] = (bf16)in[i];
}

// ---------------------------------------------------------------------------
// Async global -> LDS helpers (CDNA5 TDM-lite path, ASYNCcnt tracked)
// ---------------------------------------------------------------------------
__device__ inline void async_copy_b128(uint32_t lds_off, const void* gptr)
{
#if USE_ASYNC_LDS
    asm volatile("global_load_async_to_lds_b128 %0, %1, off"
                 :: "v"(lds_off), "v"((uint64_t)(uintptr_t)gptr)
                 : "memory");
#endif
}
__device__ inline void wait_async0()
{
#if USE_ASYNC_LDS
    asm volatile("s_wait_asynccnt 0x0" ::: "memory");
#endif
}

// ---------------------------------------------------------------------------
// Shared WMMA bf16 GEMM inner loop, double-buffered + software pipelined.
// Block tile 128x128, BK = 32, 256 threads = 8 waves, wave grid 4(M) x 2(N),
// each wave owns a 32x64 sub-tile = 2x4 WMMA 16x16 accumulators.
// A staged via GLOBAL_LOAD_ASYNC_TO_LDS_B128 (contiguous), B staged
// transposed via regs -> ds_store_b16 so fragments are contiguous ds_load_b128.
// ---------------------------------------------------------------------------
__device__ inline void gemm_loop_bf16(const bf16* __restrict__ Ag, int lda,
                                      const bf16* __restrict__ Bg, int ldb,
                                      int kTiles,
                                      bf16* __restrict__ As,   // 2*TILE halves
                                      bf16* __restrict__ Bs,   // 2*TILE halves
                                      v8f acc[2][4])
{
    const int t    = threadIdx.x;
    const int lane = t & 31;
    const int wave = t >> 5;
    const int wm   = wave >> 1;          // 0..3
    const int wn   = wave & 1;           // 0..1

    const int arow = t >> 1;             // 0..127  (A tile row)
    const int acol = (t & 1) * 16;       // 0 or 16 (A tile col base, halves)
    const int bk   = t >> 3;             // 0..31   (B tile K row)
    const int bn0  = (t & 7) * 16;       // B tile N col base

    const int khalf = (lane >> 4) * 16;  // A fragment K split per lane group
    const int kb    = (lane >> 4) * 16;  // B fragment K split per lane group
    const int lrow  = lane & 15;

    // --- staging helpers ---
    auto stageA = [&](int kt, int buf) {
        const bf16* ap = Ag + (size_t)arow * lda + kt * 32 + acol;
        bf16* dst = As + buf * TILE + arow * 32 + acol;
#if USE_ASYNC_LDS
        uint32_t l0 = (uint32_t)(uintptr_t)dst;
        async_copy_b128(l0,       ap);
        async_copy_b128(l0 + 16u, ap + 8);
#else
        *(v8bf*)dst       = *(const v8bf*)ap;
        *(v8bf*)(dst + 8) = *(const v8bf*)(ap + 8);
#endif
    };
    auto loadB = [&](int kt, v8bf& b0, v8bf& b1) {
        const bf16* bp = Bg + (size_t)(kt * 32 + bk) * ldb + bn0;
        b0 = *(const v8bf*)bp;
        b1 = *(const v8bf*)(bp + 8);
    };
    auto storeB = [&](int buf, v8bf b0, v8bf b1) {
        bf16* base = Bs + buf * TILE;
#pragma unroll
        for (int i = 0; i < 8; ++i) {
            base[(bn0 + i)     * 32 + bk] = b0[i];
            base[(bn0 + 8 + i) * 32 + bk] = b1[i];
        }
    };

    // --- prologue: stage tile 0 into buffer 0 ---
    v8bf rb0 = {}, rb1 = {};
    stageA(0, 0);
    loadB(0, rb0, rb1);
    storeB(0, rb0, rb1);
    wait_async0();
    __syncthreads();

    for (int kt = 0; kt < kTiles; ++kt) {
        const int cur  = kt & 1;
        const int nxt  = cur ^ 1;
        const bool more = (kt + 1 < kTiles);

        v8bf nb0 = {}, nb1 = {};
        if (more) {
            stageA(kt + 1, nxt);       // async, lands in other buffer
            loadB(kt + 1, nb0, nb1);   // global -> regs
        }

        const bf16* Ab = As + cur * TILE;
        const bf16* Bbuf = Bs + cur * TILE;

        // ---- A fragments (16x32): halves [khalf..+7] and [khalf+8..+15]
        v16bf afrag[2];
#pragma unroll
        for (int mi = 0; mi < 2; ++mi) {
            int row = (2 * wm + mi) * 16 + lrow;
            union { v16bf v; v8bf h[2]; } ua;
            ua.h[0] = *(const v8bf*)(Ab + row * 32 + khalf);
            ua.h[1] = *(const v8bf*)(Ab + row * 32 + khalf + 8);
            afrag[mi] = ua.v;
        }
        // ---- B fragments (32x16): 16 contiguous halves per lane
        v16bf bfrag[4];
#pragma unroll
        for (int ni = 0; ni < 4; ++ni) {
            int col = (4 * wn + ni) * 16 + lrow;
            bfrag[ni] = *(const v16bf*)(Bbuf + col * 32 + kb);
        }

#pragma unroll
        for (int mi = 0; mi < 2; ++mi)
#pragma unroll
            for (int ni = 0; ni < 4; ++ni)
                acc[mi][ni] = __builtin_amdgcn_wmma_f32_16x16x32_bf16(
                    false, afrag[mi], false, bfrag[ni],
                    (short)0, acc[mi][ni], false, false);

        if (more) {
            storeB(nxt, nb0, nb1);     // regs -> other LDS buffer
            wait_async0();             // A tile for nxt resident
        }
        __syncthreads();
    }
}

// ---------------------------------------------------------------------------
// Stage 3: gates = u @ W_in + b_in, fused sigmoid / identity / tanh epilogue.
// blockIdx.y selects the S-slice: 0 -> alpha, 1 -> b_proj, 2 -> tanh(c).
// ---------------------------------------------------------------------------
__global__ void gates_gemm_kernel(const bf16* __restrict__ U,
                                  const bf16* __restrict__ Wbf,
                                  const float* __restrict__ b_in,
                                  float* __restrict__ alpha,
                                  float* __restrict__ bpj,
                                  float* __restrict__ tcj)
{
    __shared__ __align__(64) bf16 As[2 * TILE];
    __shared__ __align__(64) bf16 Bs[2 * TILE];

    const int bm = blockIdx.x;       // 0..255
    const int bn = blockIdx.y;       // 0..2

    v8f acc[2][4];
#pragma unroll
    for (int mi = 0; mi < 2; ++mi)
#pragma unroll
        for (int ni = 0; ni < 4; ++ni) acc[mi][ni] = (v8f){};

    gemm_loop_bf16(U + (size_t)bm * 128 * Dd, Dd,
                   Wbf + bn * 128, N3, Dd / 32, As, Bs, acc);

    const int lane = threadIdx.x & 31;
    const int wave = threadIdx.x >> 5;
    const int wm = wave >> 1, wn = wave & 1;
    const int rlo = (lane >> 4) * 8;
    float* dst = (bn == 0) ? alpha : ((bn == 1) ? bpj : tcj);

#pragma unroll
    for (int mi = 0; mi < 2; ++mi)
#pragma unroll
        for (int ni = 0; ni < 4; ++ni) {
            int col = (4 * wn + ni) * 16 + (lane & 15);
            float bias = b_in[bn * 128 + col];
#pragma unroll
            for (int j = 0; j < 8; ++j) {
                int row = bm * 128 + (2 * wm + mi) * 16 + j + rlo;
                float v = acc[mi][ni][j] + bias;
                if (bn == 0)      v = 1.f / (1.f + __expf(-v));
                else if (bn == 2) v = tanhf(v);
                dst[(size_t)row * Ss + col] = v;
            }
        }
}

// ---------------------------------------------------------------------------
// Stage 4: chunked affine scan. h <- (1-a)h + a*b is h <- C*h + D.
// Pass 1: per-chunk composition (C,D). Pass 2: combine across chunks.
// Pass 3: replay chunk with correct h0 and emit y = h * tanh(c) in bf16.
// idx = (b*CHUNKS + c)*S + s keeps s fastest -> coalesced.
// ---------------------------------------------------------------------------
__global__ void scan_pass1(const float* __restrict__ alpha,
                           const float* __restrict__ bpj,
                           float* __restrict__ Cc, float* __restrict__ Dc)
{
    const int idx = blockIdx.x * blockDim.x + threadIdx.x; // 0..B*S*CHUNKS-1
    const int s = idx % Ss;
    const int c = (idx / Ss) % CHUNKS;
    const int b = idx / (Ss * CHUNKS);
    float C = 1.f, D = 0.f;
    size_t off = ((size_t)b * Tt + (size_t)c * CLEN) * Ss + s;
    for (int t = 0; t < CLEN; ++t, off += Ss) {
        float a  = alpha[off];
        float bb = bpj[off];
        float om = 1.f - a;
        C *= om;
        D  = om * D + a * bb;
    }
    Cc[idx] = C;
    Dc[idx] = D;
}

__global__ void scan_pass2(const float* __restrict__ Cc,
                           const float* __restrict__ Dc,
                           float* __restrict__ h0)
{
    const int idx = blockIdx.x * blockDim.x + threadIdx.x; // 0..B*S-1
    const int s = idx % Ss;
    const int b = idx / Ss;
    float h = 0.f;
#pragma unroll
    for (int c = 0; c < CHUNKS; ++c) {
        int j = (b * CHUNKS + c) * Ss + s;
        h0[j] = h;
        h = Cc[j] * h + Dc[j];
    }
}

__global__ void scan_pass3(const float* __restrict__ alpha,
                           const float* __restrict__ bpj,
                           const float* __restrict__ tcj,
                           const float* __restrict__ h0,
                           bf16* __restrict__ ybf)
{
    const int idx = blockIdx.x * blockDim.x + threadIdx.x; // 0..B*S*CHUNKS-1
    const int s = idx % Ss;
    const int c = (idx / Ss) % CHUNKS;
    const int b = idx / (Ss * CHUNKS);
    float h = h0[idx];
    size_t off = ((size_t)b * Tt + (size_t)c * CLEN) * Ss + s;
    for (int t = 0; t < CLEN; ++t, off += Ss) {
        float a  = alpha[off];
        float bb = bpj[off];
        h = (1.f - a) * h + a * bb;
        ybf[off] = (bf16)(h * tcj[off]);
    }
}

// ---------------------------------------------------------------------------
// Stage 5: out = x @ W_skip + y @ W_out + b_out + b_skip
// Two K-loops accumulate into the same f32 WMMA accumulators.
// ---------------------------------------------------------------------------
__global__ void out_gemm_kernel(const bf16* __restrict__ Xbf,
                                const bf16* __restrict__ Wskip,
                                const bf16* __restrict__ Ybf,
                                const bf16* __restrict__ Wout,
                                const float* __restrict__ b_out,
                                const float* __restrict__ b_skip,
                                float* __restrict__ out)
{
    __shared__ __align__(64) bf16 As[2 * TILE];
    __shared__ __align__(64) bf16 Bs[2 * TILE];

    const int bm = blockIdx.x;       // 0..255
    const int bn = blockIdx.y;       // 0..15

    v8f acc[2][4];
#pragma unroll
    for (int mi = 0; mi < 2; ++mi)
#pragma unroll
        for (int ni = 0; ni < 4; ++ni) acc[mi][ni] = (v8f){};

    // skip path: x(bf16) @ W_skip, K = 2048
    gemm_loop_bf16(Xbf + (size_t)bm * 128 * Dd, Dd,
                   Wskip + bn * 128, Dd, Dd / 32, As, Bs, acc);
    // output path: y(bf16) @ W_out, K = 128
    gemm_loop_bf16(Ybf + (size_t)bm * 128 * Ss, Ss,
                   Wout + bn * 128, Dd, Ss / 32, As, Bs, acc);

    const int lane = threadIdx.x & 31;
    const int wave = threadIdx.x >> 5;
    const int wm = wave >> 1, wn = wave & 1;
    const int rlo = (lane >> 4) * 8;

#pragma unroll
    for (int mi = 0; mi < 2; ++mi)
#pragma unroll
        for (int ni = 0; ni < 4; ++ni) {
            int gcol = bn * 128 + (4 * wn + ni) * 16 + (lane & 15);
            float bias = b_out[gcol] + b_skip[gcol];
#pragma unroll
            for (int j = 0; j < 8; ++j) {
                int row = bm * 128 + (2 * wm + mi) * 16 + j + rlo;
                out[(size_t)row * Dd + gcol] = acc[mi][ni][j] + bias;
            }
        }
}

// ---------------------------------------------------------------------------
// Host-side launch
// ---------------------------------------------------------------------------
extern "C" void kernel_launch(void* const* d_in, const int* in_sizes, int n_in,
                              void* d_out, int out_size, void* d_ws, size_t ws_size,
                              hipStream_t stream) {
    const float* x      = (const float*)d_in[0];
    const float* gamma  = (const float*)d_in[1];
    const float* beta   = (const float*)d_in[2];
    const float* W_in   = (const float*)d_in[3];
    const float* b_in   = (const float*)d_in[4];
    const float* W_out  = (const float*)d_in[5];
    const float* b_out  = (const float*)d_in[6];
    const float* W_skip = (const float*)d_in[7];
    const float* b_skip = (const float*)d_in[8];
    float* out = (float*)d_out;

    // workspace carve-up (bytes)
    char* ws = (char*)d_ws;
    size_t off = 0;
    auto carve = [&](size_t bytes) {
        char* p = ws + off;
        off += (bytes + 255) & ~(size_t)255;
        return p;
    };
    bf16*  ubf     = (bf16*)carve((size_t)Mrows * Dd * sizeof(bf16));
    bf16*  xbf     = (bf16*)carve((size_t)Mrows * Dd * sizeof(bf16));
    bf16*  winbf   = (bf16*)carve((size_t)Dd * N3 * sizeof(bf16));
    bf16*  woutbf  = (bf16*)carve((size_t)Ss * Dd * sizeof(bf16));
    bf16*  wskipbf = (bf16*)carve((size_t)Dd * Dd * sizeof(bf16));
    float* alpha   = (float*)carve((size_t)Mrows * Ss * sizeof(float));
    float* bpj     = (float*)carve((size_t)Mrows * Ss * sizeof(float));
    float* tcj     = (float*)carve((size_t)Mrows * Ss * sizeof(float));
    bf16*  ybf     = (bf16*)carve((size_t)Mrows * Ss * sizeof(bf16));
    float* Cc      = (float*)carve((size_t)Bb * Ss * CHUNKS * sizeof(float));
    float* Dc      = (float*)carve((size_t)Bb * Ss * CHUNKS * sizeof(float));
    float* h0      = (float*)carve((size_t)Bb * Ss * CHUNKS * sizeof(float));
    (void)ws_size;

    // 1. LayerNorm + bf16 conversion of activations
    ln_kernel<<<Mrows, 256, 0, stream>>>(x, gamma, beta, ubf, xbf);

    // 2. weight conversions
    cvt_bf16_kernel<<<(Dd * N3) / 256, 256, 0, stream>>>(W_in, winbf, (size_t)Dd * N3);
    cvt_bf16_kernel<<<(Ss * Dd) / 256, 256, 0, stream>>>(W_out, woutbf, (size_t)Ss * Dd);
    cvt_bf16_kernel<<<(Dd * Dd) / 256, 256, 0, stream>>>(W_skip, wskipbf, (size_t)Dd * Dd);

    // 3. gates GEMM + fused nonlinearities
    gates_gemm_kernel<<<dim3(Mrows / 128, 3), 256, 0, stream>>>(
        ubf, winbf, b_in, alpha, bpj, tcj);

    // 4. chunked recurrence scan -> y (bf16)
    scan_pass1<<<(Bb * Ss * CHUNKS) / 256, 256, 0, stream>>>(alpha, bpj, Cc, Dc);
    scan_pass2<<<(Bb * Ss) / 256, 256, 0, stream>>>(Cc, Dc, h0);
    scan_pass3<<<(Bb * Ss * CHUNKS) / 256, 256, 0, stream>>>(alpha, bpj, tcj, h0, ybf);

    // 5. output GEMM (skip + out paths fused) + biases
    out_gemm_kernel<<<dim3(Mrows / 128, Dd / 128), 256, 0, stream>>>(
        xbf, wskipbf, ybf, woutbf, b_out, b_skip, out);
}
